// PaiNet_6597069766758
// MI455X (gfx1250) — matrix operand order repaired and checked by reading, not compiled, not run on hardware
//
#include <hip/hip_runtime.h>
#include <hip/hip_bf16.h>

typedef __attribute__((ext_vector_type(16))) _Float16 v16h;
typedef __attribute__((ext_vector_type(8)))  _Float16 v8h;
typedef __attribute__((ext_vector_type(8)))  float    v8f;

#define NPTS 2048
#define BSZ  16
#define ROWS (BSZ*NPTS)   // 32768
#define KNN_K 20
#define HCOLS 512

__device__ __forceinline__ float geluf(float x) {
    return 0.5f * x * (1.0f + erff(x * 0.70710678118654752440f));
}

__device__ __forceinline__ float atomicMaxF(float* addr, float val) {
    unsigned int* ua = (unsigned int*)addr;
    unsigned int old = *ua;
    while (__uint_as_float(old) < val) {
        unsigned int assumed = old;
        old = atomicCAS(ua, assumed, __float_as_uint(val));
        if (old == assumed) break;
    }
    return __uint_as_float(old);
}

// A fragment (16-bit A 16x32 layout), source already f16 with row stride ldA (halves).
// lane = m + 16*hi holds row M=m.
// low lanes : elems 0-7 -> K=k0+0..7,   elems 8-15 -> K=k0+16..23
// high lanes: elems 0-7 -> K=k0+8..15,  elems 8-15 -> K=k0+24..31
__device__ __forceinline__ v16h load_a_frag16(const _Float16* A, int ldA, int m, int hi, int k0) {
    const _Float16* r = A + (size_t)m * ldA + k0 + hi * 8;
    v8h lo = *(const v8h*)r;
    v8h hh = *(const v8h*)(r + 16);
    return __builtin_shufflevector(lo, hh, 0, 1, 2, 3, 4, 5, 6, 7,
                                   8, 9, 10, 11, 12, 13, 14, 15);
}

// B fragment: W pre-converted f16, padded [Npad x Kpad]. Same K order as A:
// elems 0-7 -> K=k0+8*hi+0..7, elems 8-15 -> K=k0+16+8*hi+0..7.
__device__ __forceinline__ v16h load_b_frag16(const _Float16* W, int ldW, int n, int k0, int hi) {
    const _Float16* r = W + (size_t)n * ldW + k0 + hi * 8;
    v8h lo = *(const v8h*)r;
    v8h hh = *(const v8h*)(r + 16);
    return __builtin_shufflevector(lo, hh, 0, 1, 2, 3, 4, 5, 6, 7,
                                   8, 9, 10, 11, 12, 13, 14, 15);
}

// ---------------- weight convert: f32 [N x K] -> f16 [Npad x Kpad], zero padded ----------------
__global__ void convert_w_kernel(const float* __restrict__ src, _Float16* __restrict__ dst,
                                 int N, int K, int Npad, int Kpad) {
    int i = blockIdx.x * blockDim.x + threadIdx.x;
    int total = Npad * Kpad;
    if (i >= total) return;
    int n = i / Kpad, k = i - n * Kpad;
    float v = (n < N && k < K) ? src[(size_t)n * K + k] : 0.0f;
    dst[i] = (_Float16)v;
}

// ---------------- prep: x (B,3,N) -> pts (B*N,3), squared norms ----------------
__global__ void prep_kernel(const float* __restrict__ x, float* __restrict__ pts,
                            float* __restrict__ sqn) {
    int row = blockIdx.x * blockDim.x + threadIdx.x;
    if (row >= ROWS) return;
    int b = row >> 11, n = row & (NPTS - 1);
    float p0 = x[(size_t)b * 3 * NPTS + 0 * NPTS + n];
    float p1 = x[(size_t)b * 3 * NPTS + 1 * NPTS + n];
    float p2 = x[(size_t)b * 3 * NPTS + 2 * NPTS + n];
    pts[row * 3 + 0] = p0; pts[row * 3 + 1] = p1; pts[row * 3 + 2] = p2;
    sqn[row] = p0 * p0 + p1 * p1 + p2 * p2;
}

// ---------------- KNN top-20 (largest of negative sq dist), stable ties ----------------
__global__ void knn_kernel(const float* __restrict__ pts, const float* __restrict__ sqn,
                           int* __restrict__ nidx) {
    int row = blockIdx.x * blockDim.x + threadIdx.x;
    if (row >= ROWS) return;
    int b = row >> 11;
    const float* pb = pts + (size_t)(b << 11) * 3;
    const float* sb = sqn + (b << 11);
    float px = pts[row * 3], py = pts[row * 3 + 1], pz = pts[row * 3 + 2];
    float pn = sqn[row];
    float bd[KNN_K]; int bi[KNN_K];
#pragma unroll
    for (int i = 0; i < KNN_K; i++) { bd[i] = -3.4e38f; bi[i] = 0; }
    for (int m = 0; m < NPTS; m++) {
        float qx = pb[m * 3], qy = pb[m * 3 + 1], qz = pb[m * 3 + 2];
        float d = -(pn - 2.0f * (px * qx + py * qy + pz * qz) + sb[m]);
        if (d > bd[KNN_K - 1]) {
            int j = KNN_K - 1;
            while (j > 0 && bd[j - 1] < d) { bd[j] = bd[j - 1]; bi[j] = bi[j - 1]; j--; }
            bd[j] = d; bi[j] = m;
        }
    }
#pragma unroll
    for (int i = 0; i < KNN_K; i++) nidx[(size_t)row * KNN_K + i] = bi[i];
}

// ---------------- misc ----------------
__global__ void fill_kernel(float* __restrict__ p, int n, float v) {
    int i = blockIdx.x * blockDim.x + threadIdx.x;
    if (i < n) p[i] = v;
}

__global__ void bn_finalize_kernel(const float* __restrict__ sum, const float* __restrict__ sq,
                                   const float* __restrict__ g, const float* __restrict__ b,
                                   float* __restrict__ scale, float* __restrict__ shift,
                                   int ch, float invcnt) {
    int c = blockIdx.x * blockDim.x + threadIdx.x;
    if (c >= ch) return;
    float mean = sum[c] * invcnt;
    float var  = sq[c] * invcnt - mean * mean;
    float sc = g[c] * rsqrtf(var + 1e-5f);
    scale[c] = sc;
    shift[c] = b[c] - mean * sc;
}

// BN + gelu, write activated feature into f16 concat buffer H
__global__ void apply_kernel(const float* __restrict__ P, const float* __restrict__ scale,
                             const float* __restrict__ shift, _Float16* __restrict__ H,
                             int oc, int colofs, int total) {
    int i = blockIdx.x * blockDim.x + threadIdx.x;
    if (i >= total) return;
    int m = i / oc, c = i - m * oc;
    float v = geluf(P[i] * scale[c] + shift[c]);
    H[(size_t)m * HCOLS + colofs + c] = (_Float16)v;
}

__global__ void pool_final_kernel(const float* __restrict__ pmax, const float* __restrict__ psum,
                                  _Float16* __restrict__ h2) {
    int i = blockIdx.x * blockDim.x + threadIdx.x;
    if (i >= BSZ * 2048) return;
    int b = i >> 11, c = i & 2047;
    float v = (c < 1024) ? pmax[b * 1024 + c]
                         : psum[b * 1024 + (c - 1024)] * (1.0f / (float)NPTS);
    h2[i] = (_Float16)v;
}

// ---------------- PAIConv: per-WG 16 points, f32 A-tile -> f16, WMMA vs padded f16 conv_w ----------------
__global__ __launch_bounds__(256) void paiconv_kernel(
    const float* __restrict__ pts, const int* __restrict__ nidx,
    const _Float16* __restrict__ featH, int featcolofs, int use_pts,
    const float* __restrict__ Bmat, const float* __restrict__ kern,
    const float* __restrict__ mlp_w, const float* __restrict__ mlp_b,
    const _Float16* __restrict__ Wh, const float* __restrict__ conv_b,
    float* __restrict__ P, float* __restrict__ bnsum, float* __restrict__ bnsq,
    int ic, int icx, int nn, int dil, int oc, int Kpad)
{
    extern __shared__ char smem_raw[];
    float*     As   = (float*)smem_raw;              // 16*Kpad f32
    _Float16*  Ah   = (_Float16*)(As + 16 * Kpad);   // 16*Kpad f16
    float*     perm = (float*)(Ah + 16 * Kpad);      // 16*nn*nn
    float*     xnb  = perm + 16 * nn * nn;           // 16*nn*3
    float*     xrel = xnb + 16 * nn * 3;             // 16*nn*3
    float*     xrep = xrel + 16 * nn * 3;            // 16*3
    float*     ffS  = xrep + 48;                     // 16*64
    int*       nbr  = (int*)(ffS + 1024);            // 16*nn

    const int t = threadIdx.x;
    const int wgbase = blockIdx.x * 16;
    const int C = ic + icx;

    for (int s = t; s < 16 * Kpad; s += 256) As[s] = 0.0f;
    for (int s = t; s < 16 * nn; s += 256) {
        int p = s / nn, i = s - p * nn;
        int row = wgbase + p;
        int b = row >> 11;
        int g = nidx[(size_t)row * KNN_K + i * dil];
        nbr[p * nn + i] = g;
        const float* q = pts + ((size_t)(b << 11) + g) * 3;
        xnb[s * 3 + 0] = q[0]; xnb[s * 3 + 1] = q[1]; xnb[s * 3 + 2] = q[2];
    }
    __syncthreads();
    for (int s = t; s < 16 * nn; s += 256) {
        int p = s / nn, i = s - p * nn;
        float c0 = xnb[(p * nn) * 3 + 0], c1 = xnb[(p * nn) * 3 + 1], c2 = xnb[(p * nn) * 3 + 2];
        xrel[s * 3 + 0] = xnb[s * 3 + 0] - c0;
        xrel[s * 3 + 1] = xnb[s * 3 + 1] - c1;
        xrel[s * 3 + 2] = xnb[s * 3 + 2] - c2;
        if (i == 0) { xrep[p * 3 + 0] = c0; xrep[p * 3 + 1] = c1; xrep[p * 3 + 2] = c2; }
    }
    __syncthreads();
    // perm = topkmax(x_rel @ kernels + pad): softmax over i per column j, threshold, renorm
    for (int s = t; s < 16 * nn; s += 256) {
        int p = s / nn, j = s - p * nn;
        float l[KNN_K];
        float mx = -3.4e38f;
        for (int i = 0; i < nn; i++) {
            float v = xrel[(p * nn + i) * 3 + 0] * kern[0 * nn + j]
                    + xrel[(p * nn + i) * 3 + 1] * kern[1 * nn + j]
                    + xrel[(p * nn + i) * 3 + 2] * kern[2 * nn + j];
            if (i == 0 && j == 0) v += 1.0f;
            l[i] = v; mx = fmaxf(mx, v);
        }
        float sum = 0.0f;
        for (int i = 0; i < nn; i++) { l[i] = expf(l[i] - mx); sum += l[i]; }
        float inv = 1.0f / sum, csum = 0.0f;
        for (int i = 0; i < nn; i++) {
            float pv = l[i] * inv; pv = (pv > 0.1f) ? pv : 0.0f;
            l[i] = pv; csum += pv;
        }
        float rn = 1.0f / (csum + 1e-6f);
        for (int i = 0; i < nn; i++) perm[(p * nn + i) * nn + j] = l[i] * rn;
    }
    __syncthreads();
    // neighbor loop: Fourier features -> mlp -> accumulate feats^T @ perm into f32 A-tile
    for (int i = 0; i < nn; i++) {
        for (int s = t; s < 16 * 64; s += 256) {
            int p = s >> 6, m = s & 63, mm = m & 31;
            float r0 = xrel[(p * nn + i) * 3 + 0], r1 = xrel[(p * nn + i) * 3 + 1],
                  r2 = xrel[(p * nn + i) * 3 + 2];
            float dis = sqrtf(r0 * r0 + r1 * r1 + r2 * r2);
            float acc = xrep[p * 3 + 0] * Bmat[0 * 32 + mm] + xrep[p * 3 + 1] * Bmat[1 * 32 + mm]
                      + xrep[p * 3 + 2] * Bmat[2 * 32 + mm]
                      + r0 * Bmat[3 * 32 + mm] + r1 * Bmat[4 * 32 + mm] + r2 * Bmat[5 * 32 + mm]
                      + dis * Bmat[6 * 32 + mm];
            acc *= 6.28318530717958647692f;
            ffS[s] = (m < 32) ? sinf(acc) : cosf(acc);
        }
        __syncthreads();
        for (int s = t; s < 16 * C; s += 256) {
            int p = s / C, c = s - p * C;
            int row = wgbase + p, b = row >> 11;
            float v;
            if (c < ic) {
                int g = nbr[p * nn + i];
                size_t gr = (size_t)(b << 11) + g;
                v = use_pts ? pts[gr * 3 + c]
                            : (float)featH[gr * HCOLS + featcolofs + c];
            } else {
                int c2 = c - ic;
                float acc = mlp_b[c2];
                const float* wrow = mlp_w + (size_t)c2 * 64;
                const float* f = ffS + (p << 6);
                for (int q = 0; q < 64; q++) acc += f[q] * wrow[q];
                v = acc;
            }
            float* dst = As + (size_t)p * Kpad + c * nn;
            const float* pr = perm + (p * nn + i) * nn;
            for (int j = 0; j < nn; j++) dst[j] += v * pr[j];
        }
        __syncthreads();
    }
    // bulk convert A-tile f32 -> f16 (separate LDS region, single pass)
    for (int s = t; s < 16 * Kpad; s += 256) Ah[s] = (_Float16)As[s];
    __syncthreads();
    // WMMA GEMM vs Wh^T (pre-padded f16), +bias, BN stats via shfl + atomics, store pre-BN
    int w = t >> 5, lane = t & 31;
    int m = lane & 15, hi = lane >> 4;
    int ntiles = oc >> 4;
    for (int nt = w; nt < ntiles; nt += 8) {
        int n = (nt << 4) + m;
        v8f acc = {0.f, 0.f, 0.f, 0.f, 0.f, 0.f, 0.f, 0.f};
        for (int k0 = 0; k0 < Kpad; k0 += 32) {
            v16h a  = load_a_frag16(Ah, Kpad, m, hi, k0);
            v16h bf = load_b_frag16(Wh, Kpad, n, k0, hi);
            acc = __builtin_amdgcn_wmma_f32_16x16x32_f16(false, a, false, bf,
                                                         (short)0, acc, false, false);
        }
        float bias = conv_b[n];
        float s1 = 0.f, s2 = 0.f;
#pragma unroll
        for (int r = 0; r < 8; r++) { acc[r] += bias; s1 += acc[r]; s2 += acc[r] * acc[r]; }
        s1 += __shfl_xor(s1, 16, 32);
        s2 += __shfl_xor(s2, 16, 32);
        if (hi == 0) { atomicAdd(bnsum + n, s1); atomicAdd(bnsq + n, s2); }
#pragma unroll
        for (int r = 0; r < 8; r++) {
            int M = r + (hi << 3);
            P[(size_t)(wgbase + M) * oc + n] = acc[r];
        }
    }
}

// ---------------- conv5: H(f16) 32768x512 @ Wh5^T, mode 0 = stats, mode 1 = BN+gelu+pool ----------------
__global__ __launch_bounds__(256) void conv5_kernel(
    const _Float16* __restrict__ H, const _Float16* __restrict__ Wh,
    float* __restrict__ bnsum, float* __restrict__ bnsq,
    const float* __restrict__ scale, const float* __restrict__ shift,
    float* __restrict__ pmax, float* __restrict__ psum, int mode)
{
    __shared__ __align__(16) _Float16 Ah[16 * 512];
    const int t = threadIdx.x;
    const int wgbase = blockIdx.x * 16;
    for (int s = t; s < (16 * 512) / 8; s += 256) {
        int idx = s * 8;
        int p = idx >> 9, c = idx & 511;
        *(v8h*)(Ah + idx) = *(const v8h*)(H + (size_t)(wgbase + p) * HCOLS + c);
    }
    __syncthreads();
    int w = t >> 5, lane = t & 31, m = lane & 15, hi = lane >> 4;
    int b = wgbase >> 11;
    for (int nt = w; nt < 64; nt += 8) {
        int n = (nt << 4) + m;
        v8f acc = {0.f, 0.f, 0.f, 0.f, 0.f, 0.f, 0.f, 0.f};
        for (int k0 = 0; k0 < 512; k0 += 32) {
            v16h a  = load_a_frag16(Ah, 512, m, hi, k0);
            v16h bf = load_b_frag16(Wh, 512, n, k0, hi);
            acc = __builtin_amdgcn_wmma_f32_16x16x32_f16(false, a, false, bf,
                                                         (short)0, acc, false, false);
        }
        if (mode == 0) {
            float s1 = 0.f, s2 = 0.f;
#pragma unroll
            for (int r = 0; r < 8; r++) { s1 += acc[r]; s2 += acc[r] * acc[r]; }
            s1 += __shfl_xor(s1, 16, 32);
            s2 += __shfl_xor(s2, 16, 32);
            if (hi == 0) { atomicAdd(bnsum + n, s1); atomicAdd(bnsq + n, s2); }
        } else {
            float sc = scale[n], sh = shift[n];
            float mm = -3.4e38f, ss = 0.f;
#pragma unroll
            for (int r = 0; r < 8; r++) {
                float v = geluf(acc[r] * sc + sh);
                mm = fmaxf(mm, v); ss += v;
            }
            mm = fmaxf(mm, __shfl_xor(mm, 16, 32));
            ss += __shfl_xor(ss, 16, 32);
            if (hi == 0) { atomicMaxF(pmax + b * 1024 + n, mm); atomicAdd(psum + b * 1024 + n, ss); }
        }
    }
}

// ---------------- head GEMM: M=16 fits one WMMA tile; BN over batch done in-wave ----------------
__global__ void head_gemm_kernel(const _Float16* __restrict__ A, const _Float16* __restrict__ Wh,
                                 const float* __restrict__ bias,
                                 const float* __restrict__ bn_g, const float* __restrict__ bn_b,
                                 _Float16* __restrict__ out16, float* __restrict__ out32,
                                 int K, int Nvalid, int do_gelu)
{
    int lane = threadIdx.x & 31;
    int m = lane & 15, hi = lane >> 4;
    int n = blockIdx.x * 16 + m;
    bool nok = n < Nvalid;
    v8f acc = {0.f, 0.f, 0.f, 0.f, 0.f, 0.f, 0.f, 0.f};
    for (int k0 = 0; k0 < K; k0 += 32) {
        v16h a  = load_a_frag16(A, K, m, hi, k0);
        v16h bf = load_b_frag16(Wh, K, n, k0, hi);   // Wh pre-padded to tile multiple
        acc = __builtin_amdgcn_wmma_f32_16x16x32_f16(false, a, false, bf,
                                                     (short)0, acc, false, false);
    }
    if (bias != nullptr && nok) {
        float bv = bias[n];
#pragma unroll
        for (int r = 0; r < 8; r++) acc[r] += bv;
    }
    if (bn_g != nullptr) {
        float s1 = 0.f, s2 = 0.f;
#pragma unroll
        for (int r = 0; r < 8; r++) { s1 += acc[r]; s2 += acc[r] * acc[r]; }
        s1 += __shfl_xor(s1, 16, 32);
        s2 += __shfl_xor(s2, 16, 32);
        float mean = s1 * (1.0f / 16.0f);
        float var  = s2 * (1.0f / 16.0f) - mean * mean;
        float g = nok ? bn_g[n] : 1.0f, bb = nok ? bn_b[n] : 0.0f;
        float sc = g * rsqrtf(var + 1e-5f);
        float sh = bb - mean * sc;
#pragma unroll
        for (int r = 0; r < 8; r++) acc[r] = acc[r] * sc + sh;
    }
    if (do_gelu) {
#pragma unroll
        for (int r = 0; r < 8; r++) acc[r] = geluf(acc[r]);
    }
    if (nok) {
#pragma unroll
        for (int r = 0; r < 8; r++) {
            int M = r + (hi << 3);
            if (out16) out16[(size_t)M * Nvalid + n] = (_Float16)acc[r];
            else       out32[(size_t)M * Nvalid + n] = acc[r];
        }
    }
}

// ---------------- host launcher ----------------
extern "C" void kernel_launch(void* const* d_in, const int* in_sizes, int n_in,
                              void* d_out, int out_size, void* d_ws, size_t ws_size,
                              hipStream_t stream)
{
    (void)in_sizes; (void)n_in; (void)out_size; (void)ws_size;
    const float* x = (const float*)d_in[0];
    const float* conv5_w = (const float*)d_in[25];
    const float* bn5_g   = (const float*)d_in[26];
    const float* bn5_b   = (const float*)d_in[27];
    const float* lin1_w  = (const float*)d_in[28];
    const float* bn6_g   = (const float*)d_in[29];
    const float* bn6_b   = (const float*)d_in[30];
    const float* lin2_w  = (const float*)d_in[31];
    const float* lin2_b  = (const float*)d_in[32];
    const float* bn7_g   = (const float*)d_in[33];
    const float* bn7_b   = (const float*)d_in[34];
    const float* lin3_w  = (const float*)d_in[35];
    const float* lin3_b  = (const float*)d_in[36];

    const int IC[4]  = {3, 64, 64, 128};
    const int OC[4]  = {64, 64, 128, 256};
    const int DIL[4] = {1, 2, 3, 4};
    int NN[4], ICX[4], KD[4], KP[4];
    for (int li = 0; li < 4; li++) {
        NN[li]  = (KNN_K + DIL[li] - 1) / DIL[li];
        ICX[li] = (IC[li] > 3) ? IC[li] / 2 : IC[li];
        KD[li]  = (IC[li] + ICX[li]) * NN[li];
        KP[li]  = (KD[li] + 31) & ~31;
    }
    const int COL_IN[4]  = {0, 0, 64, 128};
    const int COL_OUT[4] = {0, 64, 128, 256};

    char* base = (char*)d_ws;
    size_t off = 0;
    auto carve = [&](size_t bytes) -> char* {
        char* p = base + off;
        off = (off + bytes + 255) & ~(size_t)255;
        return p;
    };
    float*     pts   = (float*)carve((size_t)ROWS * 3 * 4);
    float*     sqn   = (float*)carve((size_t)ROWS * 4);
    int*       nidx  = (int*)carve((size_t)ROWS * KNN_K * 4);
    _Float16*  H     = (_Float16*)carve((size_t)ROWS * HCOLS * 2);
    float*     P     = (float*)carve((size_t)ROWS * 256 * 4);
    float*     bnsum = (float*)carve(1024 * 4);
    float*     bnsq  = (float*)carve(1024 * 4);
    float*     scale = (float*)carve(1024 * 4);
    float*     shift = (float*)carve(1024 * 4);
    float*     pmax  = (float*)carve(BSZ * 1024 * 4);
    float*     psum  = (float*)carve(BSZ * 1024 * 4);
    _Float16*  h2    = (_Float16*)carve(BSZ * 2048 * 2);
    _Float16*  h3    = (_Float16*)carve(BSZ * 512 * 2);
    _Float16*  h4    = (_Float16*)carve(BSZ * 256 * 2);
    _Float16*  Whc   = (_Float16*)carve((size_t)256 * 960 * 2);   // per-layer conv_w, reused
    _Float16*  Wh5   = (_Float16*)carve((size_t)1024 * 512 * 2);
    _Float16*  Wl1   = (_Float16*)carve((size_t)512 * 2048 * 2);
    _Float16*  Wl2   = (_Float16*)carve((size_t)256 * 512 * 2);
    _Float16*  Wl3   = (_Float16*)carve((size_t)48 * 256 * 2);

    prep_kernel<<<(ROWS + 255) / 256, 256, 0, stream>>>(x, pts, sqn);
    knn_kernel<<<(ROWS + 255) / 256, 256, 0, stream>>>(pts, sqn, nidx);

    // one-time weight conversions (f32 -> padded f16)
    {
        int t5 = 1024 * 512;
        convert_w_kernel<<<(t5 + 255) / 256, 256, 0, stream>>>(conv5_w, Wh5, 1024, 512, 1024, 512);
        int t1 = 512 * 2048;
        convert_w_kernel<<<(t1 + 255) / 256, 256, 0, stream>>>(lin1_w, Wl1, 512, 2048, 512, 2048);
        int t2 = 256 * 512;
        convert_w_kernel<<<(t2 + 255) / 256, 256, 0, stream>>>(lin2_w, Wl2, 256, 512, 256, 512);
        int t3 = 48 * 256;
        convert_w_kernel<<<(t3 + 255) / 256, 256, 0, stream>>>(lin3_w, Wl3, 40, 256, 48, 256);
    }

    for (int li = 0; li < 4; li++) {
        const float* mlp_w  = (const float*)d_in[1 + 6 * li + 0];
        const float* mlp_b  = (const float*)d_in[1 + 6 * li + 1];
        const float* conv_w = (const float*)d_in[1 + 6 * li + 2];
        const float* conv_b = (const float*)d_in[1 + 6 * li + 3];
        const float* bn_g   = (const float*)d_in[1 + 6 * li + 4];
        const float* bn_b   = (const float*)d_in[1 + 6 * li + 5];
        const float* Bmat   = (const float*)d_in[37 + 2 * li + 0];
        const float* kern   = (const float*)d_in[37 + 2 * li + 1];
        int ic = IC[li], icx = ICX[li], nn = NN[li], dil = DIL[li];
        int oc = OC[li], Kdim = KD[li], Kpad = KP[li];

        int tw = oc * Kpad;
        convert_w_kernel<<<(tw + 255) / 256, 256, 0, stream>>>(conv_w, Whc, oc, Kdim, oc, Kpad);
        fill_kernel<<<(oc + 255) / 256, 256, 0, stream>>>(bnsum, oc, 0.0f);
        fill_kernel<<<(oc + 255) / 256, 256, 0, stream>>>(bnsq, oc, 0.0f);

        size_t lds = (size_t)4 * (16 * Kpad) + (size_t)2 * (16 * Kpad)
                   + (size_t)4 * (16 * nn * nn + 2 * 16 * nn * 3 + 48 + 1024 + 16 * nn);
        paiconv_kernel<<<ROWS / 16, 256, lds, stream>>>(
            pts, nidx, H, COL_IN[li], (li == 0) ? 1 : 0,
            Bmat, kern, mlp_w, mlp_b, Whc, conv_b,
            P, bnsum, bnsq, ic, icx, nn, dil, oc, Kpad);

        bn_finalize_kernel<<<(oc + 255) / 256, 256, 0, stream>>>(
            bnsum, bnsq, bn_g, bn_b, scale, shift, oc, 1.0f / (float)ROWS);
        int total = ROWS * oc;
        apply_kernel<<<(total + 255) / 256, 256, 0, stream>>>(
            P, scale, shift, H, oc, COL_OUT[li], total);
    }

    // conv5 pass 1: stats
    fill_kernel<<<4, 256, 0, stream>>>(bnsum, 1024, 0.0f);
    fill_kernel<<<4, 256, 0, stream>>>(bnsq, 1024, 0.0f);
    conv5_kernel<<<ROWS / 16, 256, 0, stream>>>(H, Wh5, bnsum, bnsq,
                                                nullptr, nullptr, nullptr, nullptr, 0);
    bn_finalize_kernel<<<4, 256, 0, stream>>>(bnsum, bnsq, bn5_g, bn5_b, scale, shift,
                                              1024, 1.0f / (float)ROWS);
    // conv5 pass 2: BN + gelu + pooled max/sum
    fill_kernel<<<(BSZ * 1024 + 255) / 256, 256, 0, stream>>>(pmax, BSZ * 1024, -3.4e38f);
    fill_kernel<<<(BSZ * 1024 + 255) / 256, 256, 0, stream>>>(psum, BSZ * 1024, 0.0f);
    conv5_kernel<<<ROWS / 16, 256, 0, stream>>>(H, Wh5, nullptr, nullptr,
                                                scale, shift, pmax, psum, 1);
    pool_final_kernel<<<(BSZ * 2048 + 255) / 256, 256, 0, stream>>>(pmax, psum, h2);

    // head
    head_gemm_kernel<<<512 / 16, 32, 0, stream>>>(h2, Wl1, nullptr, bn6_g, bn6_b,
                                                  h3, nullptr, 2048, 512, 1);
    head_gemm_kernel<<<256 / 16, 32, 0, stream>>>(h3, Wl2, lin2_b, bn7_g, bn7_b,
                                                  h4, nullptr, 512, 256, 1);
    head_gemm_kernel<<<48 / 16, 32, 0, stream>>>(h4, Wl3, lin3_b, nullptr, nullptr,
                                                 nullptr, (float*)d_out, 256, 40, 0);
}
